// CAE_19808389169243
// MI455X (gfx1250) — matrix operand order repaired and checked by reading, not compiled, not run on hardware
//
#include <hip/hip_runtime.h>

// ---------------------------------------------------------------------------
// Types for CDNA5 WMMA (wave32): v_wmma_f32_16x16x32_bf16
// ---------------------------------------------------------------------------
typedef __attribute__((ext_vector_type(16))) __bf16 bf16x16;
typedef __attribute__((ext_vector_type(8)))  float  f32x8;
typedef __attribute__((ext_vector_type(4)))  unsigned int u32x4;
typedef __attribute__((ext_vector_type(8)))  int i32x8;
typedef __attribute__((ext_vector_type(4)))  int i32x4;

union BF16Frag { bf16x16 v; unsigned short u[16]; };

__device__ __forceinline__ unsigned short f2bf(float f) {
  unsigned int u = __float_as_uint(f);
  u += 0x7FFFu + ((u >> 16) & 1u);   // round-to-nearest-even
  return (unsigned short)(u >> 16);
}

// B fragment: lane n holds K = {kb..kb+7} and {16+kb..16+kb+7}
__device__ __forceinline__ bf16x16 load_b_frag(const unsigned short* __restrict__ p) {
  BF16Frag r;
#pragma unroll
  for (int i = 0; i < 8; ++i) { r.u[i] = p[i]; r.u[8 + i] = p[16 + i]; }
  return r.v;
}

// A fragment: lane m holds K = {kb..kb+7} and {16+kb..16+kb+7}
__device__ __forceinline__ bf16x16 load_a_frag(const unsigned short* row, int kb) {
  BF16Frag r;
#pragma unroll
  for (int i = 0; i < 8; ++i) { r.u[i] = row[kb + i]; r.u[8 + i] = row[16 + kb + i]; }
  return r.v;
}

__device__ __forceinline__ f32x8 wmma_bf16(bf16x16 a, bf16x16 b, f32x8 c) {
  return __builtin_amdgcn_wmma_f32_16x16x32_bf16(false, a, false, b, (short)0, c, false, false);
}

// TDM: 1-D copy of `nelem` bf16 elements from global to LDS via tensor DMA.
__device__ __forceinline__ void tdm_load_1d(const void* gsrc, unsigned int lds_off,
                                            unsigned int nelem) {
  unsigned long long ga = (unsigned long long)(uintptr_t)gsrc;
  u32x4 g0;
  g0[0] = 1u;                                            // count=1 (valid descriptor)
  g0[1] = lds_off;                                       // lds_addr (bytes)
  g0[2] = (unsigned int)(ga & 0xFFFFFFFFu);              // global_addr[31:0]
  g0[3] = (unsigned int)((ga >> 32) & 0x1FFFFFFu) | (2u << 30);  // global_addr[56:32], type=2
  i32x8 g1;
  g1[0] = (int)(1u << 16);                               // data_size=1 (2 bytes), wg_mask=0
  g1[1] = (int)((nelem & 0xFFFFu) << 16);                // tensor_dim0[15:0]  (bit 48)
  g1[2] = (int)(((nelem >> 16) & 0xFFFFu) | (1u << 16)); // tensor_dim0[31:16], tensor_dim1=1
  g1[3] = (int)((nelem & 0xFFFFu) << 16);                // tile_dim0 (bit 112)
  g1[4] = 1;                                             // tile_dim1=1, tile_dim2=0
  g1[5] = (int)nelem;                                    // tensor_dim0_stride[31:0]
  g1[6] = 0;
  g1[7] = 0;
  i32x4 gz = {0, 0, 0, 0};
#if __clang_major__ >= 23
  i32x8 gz8 = {0, 0, 0, 0, 0, 0, 0, 0};
  __builtin_amdgcn_tensor_load_to_lds(g0, g1, gz, gz, gz8, 0);
#else
  __builtin_amdgcn_tensor_load_to_lds(g0, g1, gz, gz, 0);
#endif
}

// ---------------------------------------------------------------------------
// Layout/cast kernels
// ---------------------------------------------------------------------------

// x: NCHW [64,120,64,64] f32 -> halo-padded NHWC [64,66,66,128] bf16 interior
// (halo + channel pad were pre-zeroed)
__global__ void cast_x_nhwc_pad(const float* __restrict__ in, unsigned short* __restrict__ out) {
  int i = blockIdx.x * blockDim.x + threadIdx.x;
  if (i >= 64 * 64 * 64 * 128) return;
  int ci = i & 127;
  int p  = i >> 7;
  int xq = p & 63;
  int t  = p >> 6;
  int yq = t & 63;
  int b  = t >> 6;
  unsigned short v = 0;
  if (ci < 120) v = f2bf(in[(((size_t)b * 120 + ci) * 64 + yq) * 64 + xq]);
  out[(((size_t)b * 66 + yq + 1) * 66 + xq + 1) * 128 + ci] = v;
}

// conv weights OIHW [COUT][CIN][3][3] f32 -> [COUT/16][9][16][CINP] bf16 (ci zero-padded)
__global__ void cast_conv_w(const float* __restrict__ in, unsigned short* __restrict__ out,
                            int CIN, int CINP, int COUT) {
  int i = blockIdx.x * blockDim.x + threadIdx.x;
  int total = (COUT / 16) * 9 * 16 * CINP;
  if (i >= total) return;
  int ci  = i % CINP; int t = i / CINP;
  int mi  = t % 16;   t /= 16;
  int tap = t % 9;    int g = t / 9;
  int co  = g * 16 + mi;
  unsigned short v = 0;
  if (ci < CIN) v = f2bf(in[((size_t)co * CIN + ci) * 9 + tap]);
  out[i] = v;
}

// FC weights [Nsrc,Ksrc] f32 -> [Npad,Kpad] bf16, zero padded
__global__ void cast_pad_weights(const float* __restrict__ in,
                                 unsigned short* __restrict__ out,
                                 int Nsrc, int Ksrc, int Npad, int Kpad) {
  int i = blockIdx.x * blockDim.x + threadIdx.x;
  int total = Npad * Kpad;
  if (i >= total) return;
  int k = i % Kpad, nr = i / Kpad;
  unsigned short v = 0;
  if (nr < Nsrc && k < Ksrc) v = f2bf(in[(size_t)nr * Ksrc + k]);
  out[i] = v;
}

// l1w [2048,8192]: permute columns from NCHW-flatten (c*64+y*8+x) to NHWC-flatten ((y*8+x)*128+c)
__global__ void cast_l1w_perm(const float* __restrict__ in, unsigned short* __restrict__ out) {
  int i = blockIdx.x * blockDim.x + threadIdx.x;
  if (i >= 2048 * 8192) return;
  int knew = i & 8191;
  int n    = i >> 13;
  int c    = knew & 127;
  int p    = knew >> 7;             // y*8+x, 0..63
  int kold = c * 64 + p;
  out[i] = f2bf(in[(size_t)n * 8192 + kold]);
}

// ---------------------------------------------------------------------------
// Implicit-GEMM 3x3 SAME conv + bias + ReLU, halo-padded NHWC.
//   x  : [B,H+2,W+2,CINP] bf16 (1-px zero halo)   w : [COUT/16][9][16][CINP] (TDM->LDS)
//   out: [B,H,W,COUT] f32 (ReLU applied)
// block = 128 (4 waves); grid = (B*H*W/64, COUT/16)
// ---------------------------------------------------------------------------
template<int CINP, int H, int W, int COUT>
__global__ void conv3x3_wmma(const unsigned short* __restrict__ xb,
                             const unsigned short* __restrict__ wb,
                             const float* __restrict__ bias,
                             float* __restrict__ out) {
  constexpr int WBLK = 9 * 16 * CINP;          // bf16 elements of one mtile weight block
  constexpr int HP = H + 2, WP = W + 2;
  __shared__ unsigned short alds[WBLK];

  const int wave = threadIdx.x >> 5;
  const int lane = threadIdx.x & 31;
  const int lh   = lane >> 4;
  const int ln   = lane & 15;
  const int m0   = blockIdx.y * 16;

  // Stage this output-tile's weights (9 taps x 16 rows x CINP) into LDS via TDM.
  if (threadIdx.x < 32) {
    tdm_load_1d(wb + (size_t)blockIdx.y * WBLK,
                (unsigned int)(uintptr_t)(void*)alds, (unsigned int)WBLK);
    __builtin_amdgcn_s_wait_tensorcnt(0);
  }
  __syncthreads();

  const int n  = blockIdx.x * 64 + wave * 16 + ln;     // output position (b,h,w) flat
  constexpr int HW = H * W;
  const int bq = n / HW;
  const int rm = n - bq * HW;
  const int h  = rm / W;
  const int wp = rm - h * W;

  const int akb = lh ? 8 : 0;
  const int bkb = lh ? 8 : 0;

  // top-left tap of the 3x3 window in the padded image
  const unsigned short* xwin = xb + ((size_t)(bq * HP + h) * WP + wp) * CINP;

  f32x8 acc = {};
#pragma unroll
  for (int tap = 0; tap < 9; ++tap) {
    const int ky = tap / 3, kx = tap % 3;
    const unsigned short* bbase = xwin + ((size_t)ky * WP + kx) * CINP;
    const unsigned short* abase = alds + (tap * 16 + ln) * CINP;
#pragma unroll
    for (int k0 = 0; k0 < CINP; k0 += 32) {
      bf16x16 afrag = load_a_frag(abase + k0, akb);          // ds_load_b128 x2
      bf16x16 bfrag = load_b_frag(bbase + k0 + bkb);         // global_load_b128 x2
      acc = wmma_bf16(afrag, bfrag, acc);
    }
  }

  // NHWC epilogue: 8 contiguous floats per lane at n*COUT + m0 + lh*8
  const size_t obase = (size_t)n * COUT + m0 + lh * 8;
#pragma unroll
  for (int r = 0; r < 8; ++r) {
    float v = acc[r] + bias[m0 + lh * 8 + r];
    out[obase + r] = v > 0.0f ? v : 0.0f;
  }
}

// ---------------------------------------------------------------------------
// 2x2 maxpool, NHWC f32 -> NHWC bf16, optional 1-px zero halo on the output
// ---------------------------------------------------------------------------
__global__ void pool2_cast(const float* __restrict__ in,
                           unsigned short* __restrict__ out,
                           int C, int H, int W, int padOut) {
  int i = blockIdx.x * blockDim.x + threadIdx.x;
  int Ho = H >> 1, Wo = W >> 1;
  int total = 64 * C * Ho * Wo;
  if (i >= total) return;
  int c  = i % C;  int t = i / C;
  int wo = t % Wo; t /= Wo;
  int ho = t % Ho; int bb = t / Ho;
  const float* p = in + ((size_t)(bb * H + ho * 2) * W + wo * 2) * C + c;
  float v = fmaxf(fmaxf(p[0], p[C]), fmaxf(p[(size_t)W * C], p[(size_t)W * C + C]));
  int Hop = Ho + 2 * padOut, Wop = Wo + 2 * padOut;
  out[(((size_t)bb * Hop + ho + padOut) * Wop + wo + padOut) * C + c] = f2bf(v);
}

// ---------------------------------------------------------------------------
// FC layer: C[64,N] = act(A[64,K] * W[N,K]^T + bias)
// ACT 0: ReLU -> bf16 out ; ACT 1: hardtanh -> f32 out (compact [64,N])
// block = 128 (4 waves); grid = (ceil(Npad/64), 4)
// ---------------------------------------------------------------------------
template<int ACT>
__global__ void fc_wmma(const unsigned short* __restrict__ A,
                        const unsigned short* __restrict__ Wt,
                        const float* __restrict__ bias,
                        int N, int K, int Npad,
                        unsigned short* __restrict__ outb,
                        float* __restrict__ outf) {
  const int wave = threadIdx.x >> 5;
  const int lane = threadIdx.x & 31;
  const int lh   = lane >> 4;
  const int ln   = lane & 15;
  const int m0   = blockIdx.y * 16;
  const int n0   = blockIdx.x * 64 + wave * 16;
  if (n0 >= Npad) return;

  const unsigned short* arow = A  + (size_t)(m0 + ln) * K;
  const unsigned short* brow = Wt + (size_t)(n0 + ln) * K;
  const int akb = lh ? 8 : 0;
  const int bkb = lh ? 8 : 0;

  f32x8 acc = {};
  for (int k = 0; k < K; k += 32) {
    __builtin_prefetch(brow + k + 512, 0, 0);              // global_prefetch_b8 (streamed weights)
    bf16x16 afrag = load_a_frag(arow + k, akb);
    bf16x16 bfrag = load_b_frag(brow + k + bkb);
    acc = wmma_bf16(afrag, bfrag, acc);
  }

#pragma unroll
  for (int r = 0; r < 8; ++r) {
    int m = m0 + r + (lh ? 8 : 0);
    int nn = n0 + ln;
    if (nn < N) {
      float v = acc[r] + bias[nn];
      if (ACT == 0) {
        v = v > 0.0f ? v : 0.0f;
        outb[(size_t)m * N + nn] = f2bf(v);
      } else {
        v = fminf(fmaxf(v, -1.0f), 1.0f);
        outf[(size_t)m * N + nn] = v;
      }
    }
  }
}

// ---------------------------------------------------------------------------
// Utility zero fill / cube generation
// ---------------------------------------------------------------------------
__global__ void zero_f4(float4* __restrict__ p, int n) {
  int i = blockIdx.x * blockDim.x + threadIdx.x;
  int stride = gridDim.x * blockDim.x;
  float4 z = make_float4(0.f, 0.f, 0.f, 0.f);
  for (; i < n; i += stride) p[i] = z;
}

__global__ void cube_kernel(const float* __restrict__ lat,   // [64,6]
                            const float* __restrict__ xxg,   // [64,64]
                            const float* __restrict__ yyg,   // [64,64]
                            float* __restrict__ cube,        // [64,121,64,64] (pre-zeroed)
                            float* __restrict__ vout) {      // [64,64,64]
  int idx = blockIdx.x * blockDim.x + threadIdx.x;
  if (idx >= 64 * 4096) return;
  int pix = idx & 4095;
  int bb  = idx >> 12;
  const float* z = lat + bb * 6;
  float pos = atan2f(z[0], z[1]);
  float inc = z[2];
  float a   = fabsf(z[3]);
  float ah  = z[4];
  float Vh  = z[5];
  float X = xxg[pix], Y = yyg[pix];
  float cp = cosf(pos), sp = sinf(pos);
  float xt = X * cp + Y * sp;
  float yt = -X * sp + Y * cosf(inc) * cp;
  float rr = sqrtf(xt * xt + yt * yt);
  float sb = expf(-rr / a);
  float vel = sqrtf(Vh * Vh * (1.0f - (ah / rr) * atanf(rr / ah)));
  vel *= 500.0f;
  vel *= (-cosf(atan2f(xt, yt) + pos - 1.57079632679489662f) * sinf(inc));
  vout[idx] = vel;
  float bin = floorf(vel * 0.1f) + 60.0f;
  bin = fminf(fmaxf(bin, 0.0f), 120.0f);
  int c = (int)bin;
  cube[((size_t)bb * 121 + c) * 4096 + pix] = sb;
}

// ---------------------------------------------------------------------------
// Launch
// ---------------------------------------------------------------------------
extern "C" void kernel_launch(void* const* d_in, const int* in_sizes, int n_in,
                              void* d_out, int out_size, void* d_ws, size_t ws_size,
                              hipStream_t stream) {
  const float* x   = (const float*)d_in[0];
  const float* xxg = (const float*)d_in[1];
  const float* yyg = (const float*)d_in[2];
  const float* c1w = (const float*)d_in[3];  const float* c1b = (const float*)d_in[4];
  const float* c2w = (const float*)d_in[5];  const float* c2b = (const float*)d_in[6];
  const float* c3w = (const float*)d_in[7];  const float* c3b = (const float*)d_in[8];
  const float* l1w = (const float*)d_in[9];  const float* l1b = (const float*)d_in[10];
  const float* l2w = (const float*)d_in[11]; const float* l2b = (const float*)d_in[12];
  const float* l3w = (const float*)d_in[13]; const float* l3b = (const float*)d_in[14];
  const float* l4w = (const float*)d_in[15]; const float* l4b = (const float*)d_in[16];
  const float* l5w = (const float*)d_in[17]; const float* l5b = (const float*)d_in[18];

  char* ws = (char*)d_ws;
  size_t off = 0;
  auto take = [&](size_t bytes) -> char* {
    char* p = ws + off;
    off += (bytes + 255) & ~(size_t)255;
    return p;
  };
  // halo-padded NHWC activations
  const size_t XPAD_B = (size_t)64 * 66 * 66 * 128 * 2;
  const size_t P1_B   = (size_t)64 * 34 * 34 * 32 * 2;
  const size_t P2_B   = (size_t)64 * 18 * 18 * 64 * 2;
  unsigned short* xpad  = (unsigned short*)take(XPAD_B);
  unsigned short* p1h   = (unsigned short*)take(P1_B);
  unsigned short* p2h   = (unsigned short*)take(P2_B);
  unsigned short* p3    = (unsigned short*)take((size_t)64 * 8 * 8 * 128 * 2);
  unsigned short* c1wb  = (unsigned short*)take((size_t)2 * 9 * 16 * 128 * 2);
  unsigned short* c2wb  = (unsigned short*)take((size_t)4 * 9 * 16 * 32 * 2);
  unsigned short* c3wb  = (unsigned short*)take((size_t)8 * 9 * 16 * 64 * 2);
  float*          convf = (float*)take((size_t)64 * 64 * 64 * 32 * 4);             // reused
  unsigned short* l1wb  = (unsigned short*)take((size_t)2048 * 8192 * 2);
  unsigned short* l2wb  = (unsigned short*)take((size_t)1024 * 2048 * 2);
  unsigned short* l3wb  = (unsigned short*)take((size_t)512 * 1024 * 2);
  unsigned short* l4wb  = (unsigned short*)take((size_t)256 * 512 * 2);
  unsigned short* l5wb  = (unsigned short*)take((size_t)16 * 256 * 2);
  unsigned short* h1    = (unsigned short*)take((size_t)64 * 2048 * 2);
  unsigned short* h2    = (unsigned short*)take((size_t)64 * 1024 * 2);
  unsigned short* h3    = (unsigned short*)take((size_t)64 * 512 * 2);
  unsigned short* h4    = (unsigned short*)take((size_t)64 * 256 * 2);
  float*          lat   = (float*)take((size_t)64 * 6 * 4);

  float* cube = (float*)d_out;                                // 64*121*64*64
  float* vmap = (float*)d_out + (size_t)64 * 121 * 4096;

  // ---- zero halo buffers, then fill interiors ----
  zero_f4<<<2048, 256, 0, stream>>>((float4*)xpad, (int)(XPAD_B / 16));
  zero_f4<<<1024, 256, 0, stream>>>((float4*)p1h, (int)(P1_B / 16));
  zero_f4<<<1024, 256, 0, stream>>>((float4*)p2h, (int)(P2_B / 16));
  {
    int n = 64 * 64 * 64 * 128;
    cast_x_nhwc_pad<<<(n + 255) / 256, 256, 0, stream>>>(x, xpad);
  }
  cast_conv_w<<<(2 * 9 * 16 * 128 + 255) / 256, 256, 0, stream>>>(c1w, c1wb, 120, 128, 32);
  cast_conv_w<<<(4 * 9 * 16 * 32 + 255) / 256, 256, 0, stream>>>(c2w, c2wb, 32, 32, 64);
  cast_conv_w<<<(8 * 9 * 16 * 64 + 255) / 256, 256, 0, stream>>>(c3w, c3wb, 64, 64, 128);
  cast_l1w_perm<<<(2048 * 8192 + 255) / 256, 256, 0, stream>>>(l1w, l1wb);
  cast_pad_weights<<<(1024 * 2048 + 255) / 256, 256, 0, stream>>>(l2w, l2wb, 1024, 2048, 1024, 2048);
  cast_pad_weights<<<(512 * 1024 + 255) / 256, 256, 0, stream>>>(l3w, l3wb, 512, 1024, 512, 1024);
  cast_pad_weights<<<(256 * 512 + 255) / 256, 256, 0, stream>>>(l4w, l4wb, 256, 512, 256, 512);
  cast_pad_weights<<<(16 * 256 + 255) / 256, 256, 0, stream>>>(l5w, l5wb, 6, 256, 16, 256);

  // ---- conv1: [64,66,66,128] -> [64,64,64,32] ----
  conv3x3_wmma<128, 64, 64, 32><<<dim3(64 * 4096 / 64, 2), 128, 0, stream>>>(
      xpad, c1wb, c1b, convf);
  pool2_cast<<<(64 * 32 * 32 * 32 + 255) / 256, 256, 0, stream>>>(convf, p1h, 32, 64, 64, 1);

  // ---- conv2: [64,34,34,32] -> [64,32,32,64] ----
  conv3x3_wmma<32, 32, 32, 64><<<dim3(64 * 1024 / 64, 4), 128, 0, stream>>>(
      p1h, c2wb, c2b, convf);
  pool2_cast<<<(64 * 64 * 16 * 16 + 255) / 256, 256, 0, stream>>>(convf, p2h, 64, 32, 32, 1);

  // ---- conv3: [64,18,18,64] -> [64,16,16,128] ----
  conv3x3_wmma<64, 16, 16, 128><<<dim3(64 * 256 / 64, 8), 128, 0, stream>>>(
      p2h, c3wb, c3b, convf);
  pool2_cast<<<(64 * 128 * 8 * 8 + 255) / 256, 256, 0, stream>>>(convf, p3, 128, 16, 16, 0);

  // ---- MLP (p3 = [64,8192] NHWC flatten; l1w columns were permuted to match) ----
  fc_wmma<0><<<dim3(2048 / 64, 4), 128, 0, stream>>>(p3, l1wb, l1b, 2048, 8192, 2048, h1, nullptr);
  fc_wmma<0><<<dim3(1024 / 64, 4), 128, 0, stream>>>(h1, l2wb, l2b, 1024, 2048, 1024, h2, nullptr);
  fc_wmma<0><<<dim3(512 / 64, 4), 128, 0, stream>>>(h2, l3wb, l3b, 512, 1024, 512, h3, nullptr);
  fc_wmma<0><<<dim3(256 / 64, 4), 128, 0, stream>>>(h3, l4wb, l4b, 256, 512, 256, h4, nullptr);
  fc_wmma<1><<<dim3(1, 4), 128, 0, stream>>>(h4, l5wb, l5b, 6, 256, 16, nullptr, lat);

  // ---- cube rasterization ----
  {
    int n4 = 64 * 121 * 4096 / 4;
    zero_f4<<<2048, 256, 0, stream>>>((float4*)cube, n4);
    int n = 64 * 4096;
    cube_kernel<<<(n + 255) / 256, 256, 0, stream>>>(lat, xxg, yyg, cube, vmap);
  }
}
